// _ProposalLayer_56822417326431
// MI455X (gfx1250) — matrix operand order repaired and checked by reading, not compiled, not run on hardware
//
#include <hip/hip_runtime.h>
#include <hip/hip_bf16.h>
#include <math.h>

// ---------------------------------------------------------------------------
// ProposalLayer for MI455X (gfx1250, wave32, WMMA).
// Shapes: B=4, N=128, D=4096, SUB=256, H=64, P=N*(N-1)=16256, K=2048.
//
// GEMMs use v_wmma_f32_16x16x32_bf16. Each wave computes a 16x64 C strip:
// one A fragment (built from 4 aligned float4 global loads, cvt to bf16)
// feeds 4 WMMAs against pre-packed B fragments (one 32B load each).
// LSTM GEMMs collapsed from B*P rows to B*N rows via the gather structure.
// Sort + greedy NMS run one workgroup per batch image in LDS.
// ---------------------------------------------------------------------------

#define BB 4
#define NN 128
#define DD 4096
#define SUBD 256
#define HH 64
#define PP 16256           // N*(N-1)
#define TOPK 2048
#define SORTN 16384        // next pow2 >= PP
#define NMS_TH 0.7f

typedef __attribute__((ext_vector_type(16))) __bf16 v16bf;
typedef __attribute__((ext_vector_type(8)))  float  v8f;

__device__ __forceinline__ float sigmoidf_(float x) { return 1.0f / (1.0f + expf(-x)); }

// ---------------------------------------------------------------------------
// Pack B[K,N] (f32, row stride ldb==N) into WMMA-fragment order:
// Bp[kt][tn16][lane][i] = (bf16) B[(kt*32 + (lane>>4)*8 + (i&7) + (i>>3)*16) * ldb
//                                  + tn16*16 + (lane&15)]
// so a lane's 16 B values are one contiguous 32-byte run.
// ---------------------------------------------------------------------------
__global__ void pack_b_bf16(const float* __restrict__ Bm, int ldb,
                            __bf16* __restrict__ Bp, int Kdim) {
    int gid = blockIdx.x * blockDim.x + threadIdx.x;
    int nt16 = ldb >> 4;
    int total = (Kdim >> 5) * nt16 * 512;
    if (gid >= total) return;
    int i    = gid & 15;
    int lane = (gid >> 4) & 31;
    int t    = gid >> 9;                 // kt*nt16 + tn16
    int tn16 = t % nt16;
    int kt   = t / nt16;
    int k = kt * 32 + (lane >> 4) * 8 + (i & 7) + (i >> 3) * 16;
    int n = tn16 * 16 + (lane & 15);
    Bp[gid] = (__bf16)Bm[(size_t)k * ldb + n];
}

// ---------------------------------------------------------------------------
// WMMA GEMM: C[M,N] = A[M,K] * B (pre-packed) (+ bias[N]).
// grid = (M/16, N/64), block = 32 (one wave). K multiple of 32.
// Per K-step: 1 A fragment, 4 packed-B fragments, 4 v_wmma.
// ---------------------------------------------------------------------------
__global__ void gemm_wmma_bf16(const float* __restrict__ A, int lda,
                               const __bf16* __restrict__ Bp, int nt16,
                               const float* __restrict__ bias,
                               float* __restrict__ C, int ldc, int Kdim) {
    const int lane = threadIdx.x;            // 0..31
    const int tm = blockIdx.x;               // 16-row M tile
    const int nb = blockIdx.y;               // 64-col N block
    const int m    = lane & 15;
    const int kofs = (lane >> 4) * 8;        // ISA A-layout: hi lanes K+8

    const float* __restrict__ arow = A + (size_t)(tm * 16 + m) * lda;

    v8f acc[4] = {};

    for (int k0 = 0; k0 < Kdim; k0 += 32) {
        if (k0 + 32 < Kdim)
            __builtin_prefetch(arow + k0 + 32, 0, 0);   // global_prefetch_b8

        // A fragment: elements 0..7 <- K = k0+kofs..+7 ; 8..15 <- K = k0+16+kofs..+7
        float4 a0 = *(const float4*)(arow + k0 + kofs);
        float4 a1 = *(const float4*)(arow + k0 + kofs + 4);
        float4 a2 = *(const float4*)(arow + k0 + 16 + kofs);
        float4 a3 = *(const float4*)(arow + k0 + 16 + kofs + 4);
        v16bf a;
        a[0]  = (__bf16)a0.x; a[1]  = (__bf16)a0.y; a[2]  = (__bf16)a0.z; a[3]  = (__bf16)a0.w;
        a[4]  = (__bf16)a1.x; a[5]  = (__bf16)a1.y; a[6]  = (__bf16)a1.z; a[7]  = (__bf16)a1.w;
        a[8]  = (__bf16)a2.x; a[9]  = (__bf16)a2.y; a[10] = (__bf16)a2.z; a[11] = (__bf16)a2.w;
        a[12] = (__bf16)a3.x; a[13] = (__bf16)a3.y; a[14] = (__bf16)a3.z; a[15] = (__bf16)a3.w;

        const __bf16* bp = Bp + ((size_t)(k0 >> 5) * nt16 + nb * 4) * 512 + lane * 16;
        #pragma unroll
        for (int q = 0; q < 4; ++q) {
            v16bf b = *(const v16bf*)(bp + (size_t)q * 512);
            acc[q] = __builtin_amdgcn_wmma_f32_16x16x32_bf16(
                false, a, false, b, (short)0, acc[q], false, false);
        }
    }

    // C layout: acc element r -> m = r + 8*(lane>=16), n = lane&15
    #pragma unroll
    for (int q = 0; q < 4; ++q) {
        const int n = nb * 64 + q * 16 + (lane & 15);
        const float bv = bias ? bias[n] : 0.0f;
        #pragma unroll
        for (int r = 0; r < 8; ++r) {
            int mr = tm * 16 + r + 8 * (lane >> 4);
            C[(size_t)mr * ldc + n] = acc[q][r] + bv;
        }
    }
}

// ---------------------------------------------------------------------------
// Per-node elementwise LSTM (step 1 fwd + step bwd), on B*N=512 rows.
// h=c=0 input => c = sig(i)*tanh(g), h = sig(o)*tanh(c).
// ---------------------------------------------------------------------------
__global__ void node_lstm(const float* __restrict__ g1, const float* __restrict__ g3,
                          float* __restrict__ hf1, float* __restrict__ cf1,
                          float* __restrict__ hb) {
    int gid = blockIdx.x * blockDim.x + threadIdx.x;   // 512*64
    if (gid >= BB * NN * HH) return;
    int row = gid / HH, h = gid % HH;
    {
        const float* g = g1 + (size_t)row * (4 * HH);
        float c = sigmoidf_(g[h]) * tanhf(g[2 * HH + h]);
        cf1[gid] = c;
        hf1[gid] = sigmoidf_(g[3 * HH + h]) * tanhf(c);
    }
    {
        const float* g = g3 + (size_t)row * (4 * HH);
        float c = sigmoidf_(g[h]) * tanhf(g[2 * HH + h]);
        hb[gid] = sigmoidf_(g[3 * HH + h]) * tanhf(c);
    }
}

// ---------------------------------------------------------------------------
// Per-pair elementwise LSTM step 2:
// g2(s,o) = g1[o] (has b_f) + Ghf[s]; c2 = sig(f)*cf1[s] + sig(i)*tanh(g);
// h2 = sig(o)*tanh(c2).
// ---------------------------------------------------------------------------
__global__ void pair_lstm(const float* __restrict__ g1, const float* __restrict__ Ghf,
                          const float* __restrict__ cf1, float* __restrict__ h2) {
    int gid = blockIdx.x * blockDim.x + threadIdx.x;
    if (gid >= BB * PP * HH) return;
    int pr = gid / HH, hh = gid % HH;
    int b = pr / PP, p = pr % PP;
    int s = p / (NN - 1);
    int k = p % (NN - 1);
    int o = k + (k >= s ? 1 : 0);
    const float* gx = g1  + (size_t)(b * NN + o) * (4 * HH);
    const float* gh = Ghf + (size_t)(b * NN + s) * (4 * HH);
    float gi = gx[hh]          + gh[hh];
    float gf = gx[HH + hh]     + gh[HH + hh];
    float gg = gx[2 * HH + hh] + gh[2 * HH + hh];
    float go = gx[3 * HH + hh] + gh[3 * HH + hh];
    float cp = cf1[(size_t)(b * NN + s) * HH + hh];
    float c = sigmoidf_(gf) * cp + sigmoidf_(gi) * tanhf(gg);
    h2[gid] = sigmoidf_(go) * tanhf(c);
}

// ---------------------------------------------------------------------------
// Cosine score per pair: sel = selA (=h2@Wo1) + tvec[o].
// ---------------------------------------------------------------------------
__global__ void score_kernel(const float* __restrict__ selA, const float* __restrict__ tvec,
                             const float* __restrict__ nlp, float* __restrict__ scores) {
    int gid = blockIdx.x * blockDim.x + threadIdx.x;
    if (gid >= BB * PP) return;
    int b = gid / PP, p = gid % PP;
    int s = p / (NN - 1);
    int k = p % (NN - 1);
    int o = k + (k >= s ? 1 : 0);
    const float* sp = selA + (size_t)gid * HH;
    const float* tp = tvec + (size_t)(b * NN + o) * HH;
    const float* np = nlp + (size_t)b * HH;
    float dot = 0.f, na2 = 0.f, nb2 = 0.f;
    #pragma unroll 8
    for (int h = 0; h < HH; ++h) {
        float v = sp[h] + tp[h];
        float nv = np[h];
        dot += v * nv; na2 += v * v; nb2 += nv * nv;
    }
    float na = fmaxf(sqrtf(na2), 1e-6f);
    float nb = fmaxf(sqrtf(nb2), 1e-6f);
    scores[gid] = dot / (na * nb);
}

// ---------------------------------------------------------------------------
// Per-batch descending bitonic sort of (score, idx); keep top K=2048.
// One workgroup/batch, 1024 threads, 128 KB dynamic LDS.
// ---------------------------------------------------------------------------
__global__ void sort_topk(const float* __restrict__ scores,
                          int* __restrict__ order, float* __restrict__ ordsc) {
    extern __shared__ char smem[];
    float* sc  = (float*)smem;            // SORTN
    int*   idx = (int*)(sc + SORTN);      // SORTN
    const int b = blockIdx.x, tid = threadIdx.x, nt = blockDim.x;

    for (int i = tid; i < SORTN; i += nt) {
        sc[i] = (i < PP) ? scores[b * PP + i] : -INFINITY;
        idx[i] = i;
    }
    __syncthreads();

    for (int ksz = 2; ksz <= SORTN; ksz <<= 1) {
        for (int j = ksz >> 1; j > 0; j >>= 1) {
            for (int i = tid; i < SORTN; i += nt) {
                int ij = i ^ j;
                if (ij > i) {
                    bool desc = ((i & ksz) == 0);
                    float a = sc[i], c = sc[ij];
                    int ia = idx[i], ic = idx[ij];
                    bool bad = desc ? (a < c || (a == c && ia > ic))
                                    : (a > c || (a == c && ia < ic));
                    if (bad) { sc[i] = c; sc[ij] = a; idx[i] = ic; idx[ij] = ia; }
                }
            }
            __syncthreads();
        }
    }
    for (int i = tid; i < TOPK; i += nt) {
        order[b * TOPK + i] = idx[i];
        ordsc[b * TOPK + i] = sc[i];
    }
}

// ---------------------------------------------------------------------------
// Greedy NMS on min(iou_sub, iou_obj) + stable partition + output scatter.
// ---------------------------------------------------------------------------
__global__ void nms_out(const float* __restrict__ rois,
                        const int* __restrict__ order, const float* __restrict__ ordsc,
                        float* __restrict__ out9, int* __restrict__ outprop,
                        float* __restrict__ outsc) {
    extern __shared__ char smem[];
    float* bx = (float*)smem;
    float* sx1 = bx;            float* sy1 = bx + TOPK;
    float* sx2 = bx + 2*TOPK;   float* sy2 = bx + 3*TOPK;
    float* ox1 = bx + 4*TOPK;   float* oy1 = bx + 5*TOPK;
    float* ox2 = bx + 6*TOPK;   float* oy2 = bx + 7*TOPK;
    float* areaS = bx + 8*TOPK; float* areaO = bx + 9*TOPK;
    int* ip = (int*)(bx + 10*TOPK);
    int* ps   = ip;             int* po   = ip + TOPK;
    int* supp = ip + 2*TOPK;    int* keep = ip + 3*TOPK;
    int* scA  = ip + 4*TOPK;    int* scB  = ip + 5*TOPK;

    const int b = blockIdx.x, tid = threadIdx.x, nt = blockDim.x;

    for (int i = tid; i < TOPK; i += nt) {
        int ord = order[b * TOPK + i];
        int s = ord / (NN - 1);
        int k = ord % (NN - 1);
        int o = k + (k >= s ? 1 : 0);
        ps[i] = s; po[i] = o;
        const float* rs = rois + (size_t)(b * NN + s) * 5;
        const float* ro = rois + (size_t)(b * NN + o) * 5;
        sx1[i] = rs[1]; sy1[i] = rs[2]; sx2[i] = rs[3]; sy2[i] = rs[4];
        ox1[i] = ro[1]; oy1[i] = ro[2]; ox2[i] = ro[3]; oy2[i] = ro[4];
        areaS[i] = (sx2[i] - sx1[i]) * (sy2[i] - sy1[i]);
        areaO[i] = (ox2[i] - ox1[i]) * (oy2[i] - oy1[i]);
        supp[i] = 0;
    }
    __syncthreads();

    for (int i = 0; i < TOPK; ++i) {
        if (tid == 0) keep[i] = supp[i] ? 0 : 1;
        __syncthreads();
        if (keep[i]) {
            for (int j = i + 1 + tid; j < TOPK; j += nt) {
                float ltx = fmaxf(sx1[i], sx1[j]), lty = fmaxf(sy1[i], sy1[j]);
                float rbx = fminf(sx2[i], sx2[j]), rby = fminf(sy2[i], sy2[j]);
                float inter = fmaxf(rbx - ltx, 0.f) * fmaxf(rby - lty, 0.f);
                float ious = inter / (areaS[i] + areaS[j] - inter + 1e-12f);
                ltx = fmaxf(ox1[i], ox1[j]); lty = fmaxf(oy1[i], oy1[j]);
                rbx = fminf(ox2[i], ox2[j]); rby = fminf(oy2[i], oy2[j]);
                inter = fmaxf(rbx - ltx, 0.f) * fmaxf(rby - lty, 0.f);
                float iouo = inter / (areaO[i] + areaO[j] - inter + 1e-12f);
                if (fminf(ious, iouo) > NMS_TH) supp[j] = 1;
            }
        }
        __syncthreads();
    }

    for (int i = tid; i < TOPK; i += nt) scA[i] = keep[i];
    __syncthreads();
    int* src = scA; int* dst = scB;
    for (int off = 1; off < TOPK; off <<= 1) {
        for (int i = tid; i < TOPK; i += nt)
            dst[i] = src[i] + (i >= off ? src[i - off] : 0);
        __syncthreads();
        int* t = src; src = dst; dst = t;
    }

    for (int i = tid; i < TOPK; i += nt) {
        if (!keep[i]) continue;              // suppressed rows stay zero
        int dest = src[i] - 1;               // exclusive rank among kept
        float* o9 = out9 + ((size_t)b * PP + dest) * 9;
        o9[0] = (float)b;
        o9[1] = sx1[i]; o9[2] = sy1[i]; o9[3] = sx2[i]; o9[4] = sy2[i];
        o9[5] = ox1[i]; o9[6] = oy1[i]; o9[7] = ox2[i]; o9[8] = oy2[i];
        int* op = outprop + ((size_t)b * PP + dest) * 2;
        op[0] = ps[i]; op[1] = po[i];
        outsc[(size_t)b * PP + dest] = ordsc[b * TOPK + i];
    }
}

// ---------------------------------------------------------------------------
// Host-side launch.
// Inputs: 0 rois, 1 im_info(unused), 2 roi_feat, 3 nlp_features, 4 W_sub,
// 5 b_sub, 6 Wih_f, 7 Whh_f, 8 b_f, 9 Wih_b, 10 Whh_b(unused by structure),
// 11 b_b, 12 W_out, 13 b_out.
// ---------------------------------------------------------------------------
extern "C" void kernel_launch(void* const* d_in, const int* in_sizes, int n_in,
                              void* d_out, int out_size, void* d_ws, size_t ws_size,
                              hipStream_t stream) {
    const float* rois   = (const float*)d_in[0];
    const float* rfeat  = (const float*)d_in[2];
    const float* nlp    = (const float*)d_in[3];
    const float* W_sub  = (const float*)d_in[4];
    const float* b_sub  = (const float*)d_in[5];
    const float* Wih_f  = (const float*)d_in[6];
    const float* Whh_f  = (const float*)d_in[7];
    const float* b_f    = (const float*)d_in[8];
    const float* Wih_b  = (const float*)d_in[9];
    const float* b_b    = (const float*)d_in[11];
    const float* W_out  = (const float*)d_in[12];
    const float* b_out  = (const float*)d_in[13];

    // --- packed-B region (bf16), 1KB-aligned tiles, placed at ws base ---
    __bf16* pw = (__bf16*)d_ws;
    __bf16* BpWsub = pw;                         // 128*16*512 = 1048576
    __bf16* BpWihf = BpWsub + 1048576;           //   8*16*512 =   65536
    __bf16* BpWihb = BpWihf + 65536;             //   65536
    __bf16* BpWhhf = BpWihb + 65536;             //   2*16*512 =   16384
    __bf16* BpWo2  = BpWhhf + 16384;             //   2*4*512  =    4096
    __bf16* BpWo1  = BpWo2  + 4096;              //    4096
    float* ws = (float*)(BpWo1 + 4096);          // total 1204224 bf16

    float* feats = ws;                           // 512*256
    float* g1    = feats + 512 * 256;
    float* g3    = g1 + 512 * 256;
    float* hf1   = g3 + 512 * 256;               // 512*64
    float* cf1   = hf1 + 512 * 64;
    float* hb    = cf1 + 512 * 64;
    float* Ghf   = hb + 512 * 64;                // 512*256
    float* tvec  = Ghf + 512 * 256;              // 512*64
    float* h2    = tvec + 512 * 64;              // 65024*64
    float* selA  = h2 + (size_t)BB * PP * HH;    // 65024*64
    float* scores = selA + (size_t)BB * PP * HH; // 65024
    int*   order = (int*)(scores + BB * PP);     // 4*2048
    float* ordsc = (float*)(order + BB * TOPK);

    float* out9   = (float*)d_out;
    int*   oprop  = (int*)(out9 + (size_t)BB * PP * 9);
    float* osc    = (float*)(oprop + (size_t)BB * PP * 2);

    hipMemsetAsync(d_out, 0, (size_t)out_size * sizeof(float), stream);

    // 0) pre-pack all B operands into WMMA-fragment order
    pack_b_bf16<<<4096, 256, 0, stream>>>(W_sub,  SUBD,   BpWsub, DD);
    pack_b_bf16<<<256,  256, 0, stream>>>(Wih_f,  4 * HH, BpWihf, SUBD);
    pack_b_bf16<<<256,  256, 0, stream>>>(Wih_b,  4 * HH, BpWihb, SUBD);
    pack_b_bf16<<<64,   256, 0, stream>>>(Whh_f,  4 * HH, BpWhhf, HH);
    pack_b_bf16<<<16,   256, 0, stream>>>(W_out + HH * HH, HH, BpWo2, HH);
    pack_b_bf16<<<16,   256, 0, stream>>>(W_out,  HH,     BpWo1, HH);

    // 1) feats = roi_feat @ W_sub + b_sub      (512 x 4096 x 256)
    gemm_wmma_bf16<<<dim3(32, 4), 32, 0, stream>>>(rfeat, DD, BpWsub, 16, b_sub, feats, SUBD, DD);
    // 2) g1/g3 = feats @ Wih_{f,b} + b        (512 x 256 x 256)
    gemm_wmma_bf16<<<dim3(32, 4), 32, 0, stream>>>(feats, SUBD, BpWihf, 16, b_f, g1, 4 * HH, SUBD);
    gemm_wmma_bf16<<<dim3(32, 4), 32, 0, stream>>>(feats, SUBD, BpWihb, 16, b_b, g3, 4 * HH, SUBD);
    // 3) node-level LSTM gates -> hf1, cf1, hb
    node_lstm<<<128, 256, 0, stream>>>(g1, g3, hf1, cf1, hb);
    // 4) Ghf = hf1 @ Whh_f                    (512 x 64 x 256)
    gemm_wmma_bf16<<<dim3(32, 4), 32, 0, stream>>>(hf1, HH, BpWhhf, 16, nullptr, Ghf, 4 * HH, HH);
    //    tvec = hb @ W_out[H:,:] + b_out      (512 x 64 x 64)
    gemm_wmma_bf16<<<dim3(32, 1), 32, 0, stream>>>(hb, HH, BpWo2, 4, b_out, tvec, HH, HH);
    // 5) per-pair LSTM step 2 -> h2           (65024 x 64)
    pair_lstm<<<16256, 256, 0, stream>>>(g1, Ghf, cf1, h2);
    // 6) selA = h2 @ W_out[:H,:]              (65024 x 64 x 64)
    gemm_wmma_bf16<<<dim3(4064, 1), 32, 0, stream>>>(h2, HH, BpWo1, 4, nullptr, selA, HH, HH);
    // 7) cosine scores
    score_kernel<<<254, 256, 0, stream>>>(selA, tvec, nlp, scores);
    // 8) per-batch top-K sort (bitonic in 128KB LDS)
    sort_topk<<<BB, 1024, SORTN * 8, stream>>>(scores, order, ordsc);
    // 9) greedy NMS + stable partition + scatter
    nms_out<<<BB, 1024, (10 * TOPK) * 4 + (6 * TOPK) * 4, stream>>>(
        rois, order, ordsc, out9, oprop, osc);
}
